// GraphEncoder_25ChanUpSampleDecoder_10823317586530
// MI455X (gfx1250) — compile-verified, matrix operand order checked by reading, not run on hardware
//
#include <hip/hip_runtime.h>
#include <hip/hip_bf16.h>
#include <math.h>

// ---------------------------------------------------------------------------
// GraphEncoder + 25-chan upsample decoder for MI455X (gfx1250, wave32, WMMA)
// Heavy GEMMs (GNN 32768x1024x3072, obj 8192x1024x128, read 64x1024x2048,
// fc 64x2688x1024) on v_wmma_f32_16x16x32_f16 with 64x128 block tiles:
// 8 waves, each wave = 16Mx64N strip = 4 WMMA per 32-wide K step (one A-frag
// reused across 4 B-frags). GNN gather is template-specialized so the segment
// pointer mux is hoisted out of the inner loop. Decoder convs (25ch) are
// direct f32 VALU (bad 16x16 tile fit). All pads zeroed -> exact.
// ---------------------------------------------------------------------------

typedef __attribute__((ext_vector_type(16))) _Float16 v16h;
typedef __attribute__((ext_vector_type(8)))  _Float16 v8h;
typedef __attribute__((ext_vector_type(8)))  float    v8f;

#define B_     64
#define NO_    128
#define NR_    512
#define RNNP   1024     // RNN=1000 padded to 1024
#define LDST   40       // LDS row stride in halves (bank-conflict padding)

// Build a 16x32 f16 WMMA fragment from a row-major LDS tile (stride LDST).
// lane&15 selects row(A)/col(B); lane>>4 selects K-half {0..7,16..23} vs {8..15,24..31}.
__device__ __forceinline__ v16h frag16(const _Float16* p) {
  v8h x = *(const v8h*)p;
  v8h y = *(const v8h*)(p + 16);
  v16h r;
#pragma unroll
  for (int i = 0; i < 8; ++i) { r[i] = x[i]; r[i + 8] = y[i]; }
  return r;
}

// One 32-wide K step: stage A(64x32)/B(128x32) tiles in LDS, 4 WMMAs per wave.
__device__ __forceinline__ void gemm_step(const _Float16* __restrict__ aSrc,
                                          const _Float16* __restrict__ bSrc,
                                          _Float16* lA, _Float16* lB,
                                          int rowA, int chunkA, int rowB, int chunkB,
                                          const _Float16* laFrag, const _Float16* lbFrag,
                                          v8f* c) {
  v8h av  = *(const v8h*)aSrc;
  v8h bv0 = *(const v8h*)bSrc;
  v8h bv1 = *(const v8h*)(bSrc + 8);
  __builtin_prefetch(aSrc + 512, 0, 0);   // global_prefetch_b8: A stream ahead
  __builtin_prefetch(bSrc + 512, 0, 0);   // global_prefetch_b8: B stream ahead
  __syncthreads();
  *(v8h*)&lA[rowA * LDST + chunkA]     = av;
  *(v8h*)&lB[rowB * LDST + chunkB]     = bv0;
  *(v8h*)&lB[rowB * LDST + chunkB + 8] = bv1;
  __syncthreads();
  v16h af = frag16(laFrag);
#pragma unroll
  for (int j = 0; j < 4; ++j) {
    v16h bf = frag16(lbFrag + j * 16 * LDST);
    c[j] = __builtin_amdgcn_wmma_f32_16x16x32_f16(false, af, false, bf, (short)0, c[j], false, false);
  }
}

// ---------------------------------------------------------------------------
// WMMA GEMM: C(MxNp) = epilogue( A(MxKp,f16) * BT(NpxKp,f16)^T )
// MODE==1: A rows gathered on the fly: k-seg0 = obj_vecs[src], seg1 = rela,
//          seg2 = obj_vecs[dst] (each segment 1024 halves, zero-padded).
// Block: 256 thr = 8 waves as 4(M)x2(N); tile 64x128.
// ---------------------------------------------------------------------------
template <int MODE>
__global__ void k_gemm_wmma(const _Float16* __restrict__ A,
                            const _Float16* __restrict__ BT,
                            const float* __restrict__ bias, int biasN,
                            const float* __restrict__ rowMask,
                            float* __restrict__ outF,
                            _Float16* __restrict__ outH,
                            int Np, int Kp, int doRelu,
                            const _Float16* __restrict__ objvH,
                            const _Float16* __restrict__ relavH,
                            const int* __restrict__ edges) {
  __shared__ _Float16 lA[64 * LDST];
  __shared__ _Float16 lB[128 * LDST];

  const int t    = threadIdx.x;
  const int wave = t >> 5, lane = t & 31;
  const int wm   = wave >> 1;           // 0..3 -> 16-wide M sub-tile
  const int wn   = wave & 1;            // 0..1 -> 64-wide N strip
  const int mBase = blockIdx.x * 64;
  const int nBase = blockIdx.y * 128;

  const int rowA   = t >> 2;            // 0..63
  const int chunkA = (t & 3) * 8;       // 0,8,16,24
  const int rowB   = t >> 1;            // 0..127
  const int chunkB = (t & 1) * 16;      // 0,16

  const int fr = lane & 15;
  const int kh = (lane >> 4) * 8;
  const _Float16* laFrag = &lA[(wm * 16 + fr) * LDST + kh];
  const _Float16* lbFrag = &lB[(wn * 64 + fr) * LDST + kh];
  const _Float16* btRow  = BT + (size_t)(nBase + rowB) * Kp;

  v8f z = {0.f,0.f,0.f,0.f,0.f,0.f,0.f,0.f};
  v8f c[4] = {z, z, z, z};

  if (MODE == 1) {
    const int r = mBase + rowA;         // edge row this thread stages
    const int b = r >> 9;               // NR_ = 512
    const int s = edges[r * 2 + 0];
    const int o = edges[r * 2 + 1];
    const _Float16* seg0 = objvH  + (size_t)(b * NO_ + s) * RNNP + chunkA;
    const _Float16* seg1 = relavH + (size_t)r * RNNP + chunkA;
    const _Float16* seg2 = objvH  + (size_t)(b * NO_ + o) * RNNP + chunkA;
#pragma unroll 1
    for (int kk = 0; kk < 1024; kk += 32)
      gemm_step(seg0 + kk, btRow + kk + chunkB, lA, lB, rowA, chunkA, rowB, chunkB, laFrag, lbFrag, c);
#pragma unroll 1
    for (int kk = 0; kk < 1024; kk += 32)
      gemm_step(seg1 + kk, btRow + 1024 + kk + chunkB, lA, lB, rowA, chunkA, rowB, chunkB, laFrag, lbFrag, c);
#pragma unroll 1
    for (int kk = 0; kk < 1024; kk += 32)
      gemm_step(seg2 + kk, btRow + 2048 + kk + chunkB, lA, lB, rowA, chunkA, rowB, chunkB, laFrag, lbFrag, c);
  } else {
    const _Float16* aRow = A + (size_t)(mBase + rowA) * Kp + chunkA;
#pragma unroll 1
    for (int kb = 0; kb < Kp; kb += 32)
      gemm_step(aRow + kb, btRow + kb + chunkB, lA, lB, rowA, chunkA, rowB, chunkB, laFrag, lbFrag, c);
  }

  // C layout: lane fr = column, VGPR v = row v + 8*(lane>>4)
  const int rhi = (lane >> 4) * 8;
#pragma unroll
  for (int j = 0; j < 4; ++j) {
    const int col = nBase + wn * 64 + j * 16 + fr;
    const float bb = (col < biasN) ? bias[col] : 0.f;
#pragma unroll
    for (int v = 0; v < 8; ++v) {
      const int row = mBase + wm * 16 + v + rhi;
      float x = c[j][v] + bb;
      if (doRelu) x = fmaxf(x, 0.f);
      if (rowMask) x *= rowMask[row];
      if (outF) outF[(size_t)row * Np + col] = x;
      if (outH) outH[(size_t)row * Np + col] = (_Float16)x;
    }
  }
}

// ---------------------------------------------------------------------------
// Weight transpose + pad to f16: BT[n*Kp + k]; k split into segments of segKp
// (real segK). Zero pads -> exact zero contributions.
// ---------------------------------------------------------------------------
__global__ void k_transpose_pad(const float* __restrict__ W, _Float16* __restrict__ BT,
                                int N, int Np, int segK, int segKp, int Kp) {
  const size_t idx = (size_t)blockIdx.x * 256 + threadIdx.x;
  const size_t total = (size_t)Np * Kp;
  if (idx >= total) return;
  const int n = (int)(idx / Kp);
  const int k = (int)(idx % Kp);
  const int seg = k / segKp, off = k % segKp;
  _Float16 v = (_Float16)0.f;
  if (n < N && off < segK) v = (_Float16)W[(size_t)(seg * segK + off) * N + n];
  BT[idx] = v;
}

// obj_embed = relu(cat[relu(box@Wof+b), relu(emb[label])] @ Wcat + bcat) f16
__global__ void k_obj_embed(const int* __restrict__ labels, const float* __restrict__ box,
                            const float* __restrict__ emb,
                            const float* __restrict__ wof, const float* __restrict__ bof,
                            const float* __restrict__ wcat, const float* __restrict__ bcat,
                            _Float16* __restrict__ out) {
  __shared__ float cat[256];
  const int ob = blockIdx.x, t = threadIdx.x;
  const int label = labels[ob];
  float e = fmaxf(emb[label * 128 + t], 0.f);
  float f = bof[t];
#pragma unroll
  for (int i = 0; i < 5; ++i) f += box[ob * 5 + i] * wof[i * 128 + t];
  cat[t] = fmaxf(f, 0.f);
  cat[128 + t] = e;
  __syncthreads();
  float acc = bcat[t];
#pragma unroll 8
  for (int i = 0; i < 256; ++i) acc += cat[i] * wcat[i * 128 + t];
  out[(size_t)ob * 128 + t] = (_Float16)fmaxf(acc, 0.f);
}

// rela_vecs = relu(rela_feats(NRx8) @ w_rela(8x1000)) -> f16 (padded to 1024)
__global__ void k_rela_vecs(const float* __restrict__ rf, const float* __restrict__ w,
                            const float* __restrict__ b, _Float16* __restrict__ out) {
  const int r = blockIdx.x;
  const int n = blockIdx.y * 256 + threadIdx.x;    // 0..1023
  _Float16 v = (_Float16)0.f;
  if (n < 1000) {
    float acc = b[n];
#pragma unroll
    for (int i = 0; i < 8; ++i) acc += rf[r * 8 + i] * w[i * 1000 + n];
    v = (_Float16)fmaxf(acc, 0.f);
  }
  out[(size_t)r * RNNP + n] = v;
}

// attention pooling: softmax(tanh(F)@w+b) masked+renorm, weighted sum of F
__global__ void k_attn(const _Float16* __restrict__ feats, const float* __restrict__ mask,
                       const float* __restrict__ w, const float* __restrict__ b1,
                       float* __restrict__ out, int N) {
  __shared__ float sd[512];
  const int b = blockIdx.x, t = threadIdx.x;
  for (int n = t; n < N; n += 256) {
    const _Float16* f = feats + ((size_t)b * N + n) * RNNP;
    float acc = b1[0];
    for (int d = 0; d < 1000; ++d) acc += tanhf((float)f[d]) * w[d];
    sd[n] = acc;
  }
  __syncthreads();
  if (t == 0) {
    float mx = -1e30f;
    for (int n = 0; n < N; ++n) mx = fmaxf(mx, sd[n]);
    float s1 = 0.f;
    for (int n = 0; n < N; ++n) { float e = expf(sd[n] - mx); sd[n] = e; s1 += e; }
    float s2 = 0.f;
    for (int n = 0; n < N; ++n) { float wv = sd[n] / s1 * mask[b * N + n]; sd[n] = wv; s2 += wv; }
    const float inv = 1.f / s2;
    for (int n = 0; n < N; ++n) sd[n] *= inv;
  }
  __syncthreads();
  for (int d = t; d < 1000; d += 256) {
    float acc = 0.f;
    for (int n = 0; n < N; ++n) acc += sd[n] * (float)feats[((size_t)b * N + n) * RNNP + d];
    out[b * 1000 + d] = acc;
  }
}

// A_read (64 x 2048 f16) = [a_obj pad1024 | a_rel pad1024]
__global__ void k_build_aread(const float* __restrict__ aObj, const float* __restrict__ aRel,
                              _Float16* __restrict__ A) {
  const int idx = blockIdx.x * 256 + threadIdx.x;
  if (idx >= 64 * 2048) return;
  const int b = idx >> 11, c = idx & 2047;
  const int seg = c >> 10, off = c & 1023;
  float v = 0.f;
  if (off < 1000) v = seg ? aRel[b * 1000 + off] : aObj[b * 1000 + off];
  A[idx] = (_Float16)v;
}

// fused upsample(x2) + "full" 3x3 conv-transpose:
// y[b,co,i,j] = bias[co] + sum_{ci,p,q} xup[b,ci,i-p,j-q] * w[ci,co,p,q]
__global__ void k_up2convT(const float* __restrict__ x, const float* __restrict__ w,
                           const float* __restrict__ bias, float* __restrict__ y,
                           int Bn, int Cin, int Cout, int H, int W, int act) {
  const int OH = 2 * H + 2, OW = 2 * W + 2;
  const size_t total = (size_t)Bn * Cout * OH * OW;
  const size_t id = (size_t)blockIdx.x * 256 + threadIdx.x;
  if (id >= total) return;
  const int j = (int)(id % OW);
  size_t r1 = id / OW;
  const int i = (int)(r1 % OH);
  size_t r2 = r1 / OH;
  const int co = (int)(r2 % Cout);
  const int b = (int)(r2 / Cout);
  float acc = bias[co];
  for (int ci = 0; ci < Cin; ++ci) {
    const float* xp = x + ((size_t)b * Cin + ci) * H * W;
    const float* wp = w + ((size_t)ci * Cout + co) * 9;
#pragma unroll
    for (int p = 0; p < 3; ++p) {
      const int ii = i - p;
      if (ii < 0 || ii >= 2 * H) continue;
      const float* xr = xp + (ii >> 1) * W;
#pragma unroll
      for (int q = 0; q < 3; ++q) {
        const int jj = j - q;
        if (jj < 0 || jj >= 2 * W) continue;
        acc += xr[jj >> 1] * wp[p * 3 + q];
      }
    }
  }
  acc = act ? (1.f / (1.f + expf(-acc))) : fmaxf(acc, 0.f);
  y[id] = acc;
}

// ---------------------------------------------------------------------------
extern "C" void kernel_launch(void* const* d_in, const int* in_sizes, int n_in,
                              void* d_out, int out_size, void* d_ws, size_t ws_size,
                              hipStream_t stream) {
  const int*   obj_labels = (const int*)  d_in[0];
  const float* obj_masks  = (const float*)d_in[1];
  const float* rela_masks = (const float*)d_in[2];
  const int*   rela_edges = (const int*)  d_in[3];
  const float* rela_feats = (const float*)d_in[4];
  const float* box_feats  = (const float*)d_in[5];
  const float* emb_table  = (const float*)d_in[6];
  const float* w_objfeat  = (const float*)d_in[7];
  const float* b_objfeat  = (const float*)d_in[8];
  const float* w_cat      = (const float*)d_in[9];
  const float* b_cat      = (const float*)d_in[10];
  const float* w_obj      = (const float*)d_in[11];
  const float* b_obj      = (const float*)d_in[12];
  const float* w_rela     = (const float*)d_in[13];
  const float* b_rela     = (const float*)d_in[14];
  const float* w_gnn      = (const float*)d_in[15];
  const float* b_gnn      = (const float*)d_in[16];
  const float* w_att_obj  = (const float*)d_in[17];
  const float* b_att_obj  = (const float*)d_in[18];
  const float* w_att_rela = (const float*)d_in[19];
  const float* b_att_rela = (const float*)d_in[20];
  const float* w_read     = (const float*)d_in[21];
  const float* b_read     = (const float*)d_in[22];
  const float* w_fc       = (const float*)d_in[23];
  const float* b_fc       = (const float*)d_in[24];
  const float* wt1 = (const float*)d_in[25]; const float* bt1 = (const float*)d_in[26];
  const float* wt2 = (const float*)d_in[27]; const float* bt2 = (const float*)d_in[28];
  const float* wt3 = (const float*)d_in[29]; const float* bt3 = (const float*)d_in[30];
  const float* wt4 = (const float*)d_in[31]; const float* bt4 = (const float*)d_in[32];

  char* ws = (char*)d_ws;
  size_t cur = 0;
  auto alloc = [&](size_t bytes) -> void* {
    cur = (cur + 255) & ~(size_t)255;
    void* p = ws + cur;
    cur += bytes;
    return p;
  };

  const int M_OBJ = B_ * NO_;   // 8192
  const int M_REL = B_ * NR_;   // 32768

  _Float16* objEmbH  = (_Float16*)alloc((size_t)M_OBJ * 128 * 2);
  _Float16* objvH    = (_Float16*)alloc((size_t)M_OBJ * RNNP * 2);
  _Float16* relavH   = (_Float16*)alloc((size_t)M_REL * RNNP * 2);
  _Float16* newRelaH = (_Float16*)alloc((size_t)M_REL * RNNP * 2);
  _Float16* BT_obj   = (_Float16*)alloc((size_t)RNNP * 128 * 2);
  _Float16* BT_gnn   = (_Float16*)alloc((size_t)RNNP * 3072 * 2);
  _Float16* BT_read  = (_Float16*)alloc((size_t)1024 * 2048 * 2);
  _Float16* BT_fc    = (_Float16*)alloc((size_t)2688 * 1024 * 2);
  float*    aObj     = (float*)   alloc((size_t)B_ * 1000 * 4);
  float*    aRel     = (float*)   alloc((size_t)B_ * 1000 * 4);
  _Float16* A_read   = (_Float16*)alloc((size_t)B_ * 2048 * 2);
  _Float16* encH     = (_Float16*)alloc((size_t)B_ * 1024 * 2);
  float*    x0       = (float*)   alloc((size_t)B_ * 2688 * 4);
  float*    x1       = (float*)   alloc((size_t)B_ * 25 * 30 * 14 * 4);
  float*    x2       = (float*)   alloc((size_t)B_ * 25 * 62 * 30 * 4);
  float*    x3       = (float*)   alloc((size_t)B_ * 25 * 126 * 62 * 4);
  (void)alloc(4096);  // prefetch-overrun guard

  float* encOut = (float*)d_out;                     // (64,1024)
  float* xOut   = (float*)d_out + (size_t)B_ * 1024; // (64,25,254,126)

  auto cdiv = [](size_t a, size_t b) { return (unsigned)((a + b - 1) / b); };

  // 1) weight packs
  k_transpose_pad<<<cdiv((size_t)RNNP * 128, 256), 256, 0, stream>>>(w_obj,  BT_obj,  1000, RNNP, 128, 128, 128);
  k_transpose_pad<<<cdiv((size_t)RNNP * 3072, 256), 256, 0, stream>>>(w_gnn,  BT_gnn,  1000, RNNP, 1000, 1024, 3072);
  k_transpose_pad<<<cdiv((size_t)1024 * 2048, 256), 256, 0, stream>>>(w_read, BT_read, 1024, 1024, 1000, 1024, 2048);
  k_transpose_pad<<<cdiv((size_t)2688 * 1024, 256), 256, 0, stream>>>(w_fc,   BT_fc,   2688, 2688, 1024, 1024, 1024);

  // 2) front-end MLPs
  k_obj_embed<<<M_OBJ, 128, 0, stream>>>(obj_labels, box_feats, emb_table,
                                         w_objfeat, b_objfeat, w_cat, b_cat, objEmbH);
  k_rela_vecs<<<dim3(M_REL, 4), 256, 0, stream>>>(rela_feats, w_rela, b_rela, relavH);

  // 3) obj_vecs = relu(objEmb @ w_obj)  [WMMA 8192x1024x128]
  k_gemm_wmma<0><<<dim3(M_OBJ / 64, RNNP / 128), 256, 0, stream>>>(
      objEmbH, BT_obj, b_obj, 1000, nullptr, nullptr, objvH,
      RNNP, 128, /*relu*/1, nullptr, nullptr, nullptr);

  // 4) GNN: new_rela = relu([s|rela|o] @ w_gnn) * mask  [WMMA 32768x1024x3072]
  k_gemm_wmma<1><<<dim3(M_REL / 64, RNNP / 128), 256, 0, stream>>>(
      nullptr, BT_gnn, b_gnn, 1000, rela_masks, nullptr, newRelaH,
      RNNP, 3072, /*relu*/1, objvH, relavH, rela_edges);

  // 5) attention pooling
  k_attn<<<B_, 256, 0, stream>>>(objvH,    obj_masks,  w_att_obj,  b_att_obj,  aObj, NO_);
  k_attn<<<B_, 256, 0, stream>>>(newRelaH, rela_masks, w_att_rela, b_att_rela, aRel, NR_);

  // 6) enc = relu([a_obj|a_rel] @ w_read)  [WMMA 64x1024x2048]
  k_build_aread<<<cdiv(64 * 2048, 256), 256, 0, stream>>>(aObj, aRel, A_read);
  k_gemm_wmma<0><<<dim3(1, 1024 / 128), 256, 0, stream>>>(
      A_read, BT_read, b_read, 1024, nullptr, encOut, encH,
      1024, 2048, /*relu*/1, nullptr, nullptr, nullptr);

  // 7) x0 = enc @ w_fc (no relu)  [WMMA 64x2688x1024]
  k_gemm_wmma<0><<<dim3(1, 2688 / 128), 256, 0, stream>>>(
      encH, BT_fc, b_fc, 2688, nullptr, x0, nullptr,
      2688, 1024, /*relu*/0, nullptr, nullptr, nullptr);

  // 8) decoder: up2 + convT chain
  k_up2convT<<<cdiv((size_t)B_ * 25 * 30 * 14, 256), 256, 0, stream>>>(
      x0, wt1, bt1, x1, B_, 32, 25, 14, 6, 0);
  k_up2convT<<<cdiv((size_t)B_ * 25 * 62 * 30, 256), 256, 0, stream>>>(
      x1, wt2, bt2, x2, B_, 25, 25, 30, 14, 0);
  k_up2convT<<<cdiv((size_t)B_ * 25 * 126 * 62, 256), 256, 0, stream>>>(
      x2, wt3, bt3, x3, B_, 25, 25, 62, 30, 0);
  k_up2convT<<<cdiv((size_t)B_ * 25 * 254 * 126, 256), 256, 0, stream>>>(
      x3, wt4, bt4, xOut, B_, 25, 25, 126, 62, 1);
}
